// Qwen3Attention_16956530884940
// MI455X (gfx1250) — matrix operand order repaired and checked by reading, not compiled, not run on hardware
//
#include <hip/hip_runtime.h>

// ---------------------------------------------------------------------------
// Types
// ---------------------------------------------------------------------------
typedef __attribute__((ext_vector_type(16))) __bf16 v16bf;
typedef __attribute__((ext_vector_type(8)))  __bf16 v8bf;
typedef __attribute__((ext_vector_type(8)))  float  v8f;

// ---------------------------------------------------------------------------
// Problem constants
// ---------------------------------------------------------------------------
constexpr int Bc   = 2;
constexpr int Sc   = 2048;
constexpr int Hc   = 4096;
constexpr int NQc  = 32;
constexpr int NKVc = 8;
constexpr int Dc   = 128;
constexpr int Mrows = Bc * Sc;                 // 4096 tokens
constexpr int QKVN  = (NQc + 2 * NKVc) * Dc;   // 6144

// ---------------------------------------------------------------------------
// CDNA5 helpers
// ---------------------------------------------------------------------------
#define USE_ASYNC_LDS 1

#if USE_ASYNC_LDS && defined(__AMDGCN__) && __has_builtin(__builtin_amdgcn_global_load_async_to_lds_b128)
#define HAVE_ASYNC_LDS 1
#endif

// 16-byte global -> LDS copy (async path on gfx1250 when available)
__device__ __forceinline__ void cp16_g2s(__bf16* l, const __bf16* g) {
#if defined(HAVE_ASYNC_LDS)
  typedef int v4i_ __attribute__((vector_size(16)));
  v4i_* gv = (v4i_*)g;  // drop const + reinterpret (generic AS)
  v4i_* lv = (v4i_*)l;
  __builtin_amdgcn_global_load_async_to_lds_b128(
      (__attribute__((address_space(1))) v4i_*)gv,
      (__attribute__((address_space(3))) v4i_*)lv, 0, 0);
#else
  *(v8bf*)l = *(const v8bf*)g;
#endif
}

__device__ __forceinline__ void wait_async_lds() {
#if defined(HAVE_ASYNC_LDS)
#if __has_builtin(__builtin_amdgcn_s_wait_asynccnt)
  __builtin_amdgcn_s_wait_asynccnt(0);
#else
  asm volatile("s_wait_asynccnt 0" ::: "memory");
#endif
#endif
}

__device__ __forceinline__ void wait_ds0() {
#if defined(__AMDGCN__)
#if __has_builtin(__builtin_amdgcn_s_wait_dscnt)
  __builtin_amdgcn_s_wait_dscnt(0);
#else
  asm volatile("s_wait_dscnt 0" ::: "memory");
#endif
#endif
}

// A-matrix fragment (16x32 bf16, MxK). Lane m = lane&15, half = lane>>4.
// K chunks: [half*8 .. half*8+7] and [16+half*8 .. 16+half*8+7].
__device__ __forceinline__ v16bf load_fragA(const __bf16* base, int stride, int lane) {
  const int r = lane & 15, h = lane >> 4;
  v8bf lo = *(const v8bf*)(base + (long long)r * stride + h * 8);
  v8bf hi = *(const v8bf*)(base + (long long)r * stride + 16 + h * 8);
  return __builtin_shufflevector(lo, hi, 0, 1, 2, 3, 4, 5, 6, 7,
                                 8, 9, 10, 11, 12, 13, 14, 15);
}

// B-matrix fragment (32x16 bf16, KxN) loaded from a B^T ([N][K]) layout.
// Lane n = lane&15, half = lane>>4. K chunks: [half*8 .. half*8+7] and [16+half*8 .. 16+half*8+7].
__device__ __forceinline__ v16bf load_fragB(const __bf16* base, int stride, int lane) {
  const int r = lane & 15, h = lane >> 4;
  v8bf lo = *(const v8bf*)(base + (long long)r * stride + h * 8);
  v8bf hi = *(const v8bf*)(base + (long long)r * stride + 16 + h * 8);
  return __builtin_shufflevector(lo, hi, 0, 1, 2, 3, 4, 5, 6, 7,
                                 8, 9, 10, 11, 12, 13, 14, 15);
}

__device__ __forceinline__ v8f wmma_bf16(v16bf a, v16bf b, v8f c) {
  return __builtin_amdgcn_wmma_f32_16x16x32_bf16(false, a, false, b,
                                                 (short)0, c, false, false);
}

// ---------------------------------------------------------------------------
// Kernel 1: fp32 -> bf16 elementwise
// ---------------------------------------------------------------------------
__global__ void k_f32_to_bf16(const float* __restrict__ src,
                              __bf16* __restrict__ dst, long long n) {
  long long i = (long long)blockIdx.x * 256 + threadIdx.x;
  if (i < n) dst[i] = (__bf16)src[i];
}

// ---------------------------------------------------------------------------
// Kernel 2: transpose-convert W [K][N] fp32 -> W^T [N][K] bf16
// ---------------------------------------------------------------------------
__global__ void k_transpose_bf16(const float* __restrict__ src,
                                 __bf16* __restrict__ dst, int K, int N) {
  long long idx = (long long)blockIdx.x * 256 + threadIdx.x;
  long long total = (long long)K * N;
  if (idx >= total) return;
  int n = (int)(idx / K);
  int k = (int)(idx % K);
  dst[idx] = (__bf16)src[(long long)k * N + n];
}

// ---------------------------------------------------------------------------
// Kernel 3: WMMA bf16 GEMM:  C[M][N] = A[M][K] * B, with B given as B^T [N][K]
// Block 128x128, 256 threads = 8 waves (2x4), wave tile 64x32, K-step 32.
// ---------------------------------------------------------------------------
template <bool OUTF32>
__global__ void __launch_bounds__(256)
k_gemm(const __bf16* __restrict__ A, const __bf16* __restrict__ BT,
       void* __restrict__ Cout, int M, int N, int K) {
  __shared__ __bf16 sA[128 * 32];
  __shared__ __bf16 sB[128 * 32];

  const int t = threadIdx.x;
  const int lane = t & 31;
  const int w = t >> 5;
  const int wm = w >> 2;   // 0..1
  const int wn = w & 3;    // 0..3
  const int m0 = blockIdx.x * 128;
  const int n0 = blockIdx.y * 128;

  v8f C[4][2];
#pragma unroll
  for (int i = 0; i < 4; ++i)
#pragma unroll
    for (int j = 0; j < 2; ++j) C[i][j] = {};

  for (int k0 = 0; k0 < K; k0 += 32) {
    // stage A and B^T tiles (128 rows x 32 cols bf16 each)
    for (int c = t; c < 512; c += 256) {
      int row = c >> 2;
      int off = (c & 3) * 8;
      cp16_g2s(&sA[row * 32 + off], A + (long long)(m0 + row) * K + k0 + off);
      cp16_g2s(&sB[row * 32 + off], BT + (long long)(n0 + row) * K + k0 + off);
    }
    wait_async_lds();
    __syncthreads();

    v16bf a[4], b[2];
#pragma unroll
    for (int i = 0; i < 4; ++i)
      a[i] = load_fragA(&sA[(wm * 64 + i * 16) * 32], 32, lane);
#pragma unroll
    for (int j = 0; j < 2; ++j)
      b[j] = load_fragB(&sB[(wn * 32 + j * 16) * 32], 32, lane);

#pragma unroll
    for (int i = 0; i < 4; ++i)
#pragma unroll
      for (int j = 0; j < 2; ++j) C[i][j] = wmma_bf16(a[i], b[j], C[i][j]);

    __syncthreads();
  }

  const int nn = lane & 15;
  const int half = lane >> 4;
#pragma unroll
  for (int i = 0; i < 4; ++i)
#pragma unroll
    for (int j = 0; j < 2; ++j)
#pragma unroll
      for (int r = 0; r < 8; ++r) {
        int row = m0 + wm * 64 + i * 16 + r + 8 * half;
        int col = n0 + wn * 32 + j * 16 + nn;
        if (OUTF32)
          ((float*)Cout)[(long long)row * N + col] = C[i][j][r];
        else
          ((__bf16*)Cout)[(long long)row * N + col] = (__bf16)C[i][j][r];
      }
}

// ---------------------------------------------------------------------------
// Kernel 4: per-head RMSNorm + RoPE, scatter to head-major layouts.
//   qkv bf16 [M][6144] ; q -> qh [b][32][s][128], k -> kh [b][8][s][128],
//   v -> vT [b][8][128][s]   (transposed so PV B-fragments read contiguously)
// One block = (token, head), 128 threads = D.
// ---------------------------------------------------------------------------
__global__ void __launch_bounds__(128)
k_normrope(const __bf16* __restrict__ qkv, const float* __restrict__ qw,
           const float* __restrict__ kw, __bf16* __restrict__ qh,
           __bf16* __restrict__ kh, __bf16* __restrict__ vT) {
  __shared__ float red[128];
  __shared__ float sm[128];

  const int tkn = blockIdx.x / 48;
  const int hh = blockIdx.x % 48;
  const int d = threadIdx.x;
  const int b = tkn / Sc;
  const int s = tkn % Sc;

  float val = (float)qkv[(long long)tkn * QKVN + hh * Dc + d];

  if (hh >= 40) {  // V head: no norm, no rope, store transposed
    int kvh = hh - 40;
    vT[((long long)(b * NKVc + kvh) * Dc + d) * Sc + s] = (__bf16)val;
    return;
  }

  red[d] = val * val;
  __syncthreads();
#pragma unroll
  for (int off = 64; off > 0; off >>= 1) {
    if (d < off) red[d] += red[d + off];
    __syncthreads();
  }
  float ms = red[0] * (1.0f / 128.0f);
  float rr = rsqrtf(ms + 1e-6f);
  const float* w = (hh < 32) ? qw : kw;
  float nv = val * rr * w[d];
  sm[d] = nv;
  __syncthreads();

  int i = d & 63;
  // inv_freq = theta^(-i/64), theta = 1e6
  float freq = __expf(-(float)i * (13.815510557964274f / 64.0f));
  float ang = (float)s * freq;
  float sn, cs;
  __sincosf(ang, &sn, &cs);
  float x1 = sm[i];
  float x2 = sm[i + 64];
  float out = (d < 64) ? (x1 * cs - x2 * sn) : (x2 * cs + x1 * sn);

  if (hh < 32)
    qh[((long long)(b * NQc + hh) * Sc + s) * Dc + d] = (__bf16)out;
  else
    kh[((long long)(b * NKVc + (hh - 32)) * Sc + s) * Dc + d] = (__bf16)out;
}

// ---------------------------------------------------------------------------
// Kernel 5: causal flash attention (GQA, G=4).
// Block = 256 threads = 8 waves; each wave owns a 16-row Q tile.
// Grid = B * NQ * (S/128). Per 32 KV positions: 8 WMMA (QK^T) + 8 WMMA (PV).
// ---------------------------------------------------------------------------
__global__ void __launch_bounds__(256)
k_attn(const __bf16* __restrict__ qh, const __bf16* __restrict__ kh,
       const __bf16* __restrict__ vT, __bf16* __restrict__ attn) {
  __shared__ __bf16 sP[8 * 16 * 32];  // per-wave 16x32 P tile

  const int bid = blockIdx.x;
  const int qt = bid & 15;
  const int h = (bid >> 4) & 31;
  const int b = bid >> 9;
  const int w = threadIdx.x >> 5;
  const int lane = threadIdx.x & 31;
  const int nn = lane & 15;
  const int half = lane >> 4;
  const int q0 = qt * 128 + w * 16;
  const int kvh = h >> 2;

  const __bf16* qbase = qh + ((long long)(b * NQc + h) * Sc + q0) * Dc;
  const __bf16* kbase = kh + (long long)(b * NKVc + kvh) * Sc * Dc;
  const __bf16* vbase = vT + (long long)(b * NKVc + kvh) * Dc * Sc;
  __bf16* pb = sP + w * 512;

  // Q fragments: 4 chunks over D=128, held in registers for the whole row
  v16bf qf[4];
#pragma unroll
  for (int c = 0; c < 4; ++c) qf[c] = load_fragA(qbase + c * 32, Dc, lane);

  v8f O[8];
#pragma unroll
  for (int cf = 0; cf < 8; ++cf) O[cf] = {};
  float rmax[8], rsum[8];
#pragma unroll
  for (int r = 0; r < 8; ++r) { rmax[r] = -3.0e38f; rsum[r] = 0.0f; }

  const float scale = 0.08838834764831845f;   // 1/sqrt(128)
  const float L2E = 1.4426950408889634f;
  const int nch = (q0 + 16 + 31) >> 5;

  for (int ch = 0; ch < nch; ++ch) {
    const int kb = ch * 32;

    // scores: two 16x16 fragments covering kv columns [kb, kb+32)
    v8f s0 = {}, s1 = {};
#pragma unroll
    for (int c = 0; c < 4; ++c) {
      v16bf b0 = load_fragB(kbase + (long long)kb * Dc + c * 32, Dc, lane);
      v16bf b1 = load_fragB(kbase + (long long)(kb + 16) * Dc + c * 32, Dc, lane);
      s0 = wmma_bf16(qf[c], b0, s0);
      s1 = wmma_bf16(qf[c], b1, s1);
    }

    // online softmax update (row stats replicated across each 16-lane half)
    float corr[8];
#pragma unroll
    for (int r = 0; r < 8; ++r) {
      const int row = q0 + r + 8 * half;
      float a0 = (kb + nn <= row) ? s0[r] * scale : -3.0e38f;
      float a1 = (kb + 16 + nn <= row) ? s1[r] * scale : -3.0e38f;
      float mx = fmaxf(a0, a1);
#pragma unroll
      for (int off = 1; off < 16; off <<= 1)
        mx = fmaxf(mx, __shfl_xor(mx, off, 32));
      float mnew = fmaxf(rmax[r], mx);
      float c0 = exp2f((rmax[r] - mnew) * L2E);
      rmax[r] = mnew;
      float p0 = exp2f((a0 - mnew) * L2E);
      float p1 = exp2f((a1 - mnew) * L2E);
      float ps = p0 + p1;
#pragma unroll
      for (int off = 1; off < 16; off <<= 1) ps += __shfl_xor(ps, off, 32);
      rsum[r] = rsum[r] * c0 + ps;
      corr[r] = c0;
      const int m = r + 8 * half;
      pb[m * 32 + nn] = (__bf16)p0;
      pb[m * 32 + 16 + nn] = (__bf16)p1;
    }

    // rescale running output
#pragma unroll
    for (int cf = 0; cf < 8; ++cf)
#pragma unroll
      for (int r = 0; r < 8; ++r) O[cf][r] *= corr[r];

    // P (C-layout) -> A-layout via wave-private LDS tile
    wait_ds0();
    v16bf pf = load_fragA(pb, 32, lane);

    // O += P @ V  (V^T rows are d, contiguous in s: direct B fragments)
#pragma unroll
    for (int cf = 0; cf < 8; ++cf) {
      v16bf vfrag = load_fragB(vbase + (long long)(cf * 16) * Sc + kb, Sc, lane);
      O[cf] = wmma_bf16(pf, vfrag, O[cf]);
    }
  }

  // normalize and store bf16 to attn [M][NQ*D]
  const long long orow = (long long)(b * Sc + q0);
#pragma unroll
  for (int cf = 0; cf < 8; ++cf)
#pragma unroll
    for (int r = 0; r < 8; ++r) {
      const int m = r + 8 * half;
      float v = O[cf][r] / rsum[r];
      attn[(orow + m) * (long long)(NQc * Dc) + h * Dc + cf * 16 + nn] =
          (__bf16)v;
    }
}

// ---------------------------------------------------------------------------
// Host launcher
// ---------------------------------------------------------------------------
extern "C" void kernel_launch(void* const* d_in, const int* in_sizes, int n_in,
                              void* d_out, int out_size, void* d_ws,
                              size_t ws_size, hipStream_t stream) {
  const float* x  = (const float*)d_in[0];
  const float* wq = (const float*)d_in[1];
  const float* wk = (const float*)d_in[2];
  const float* wv = (const float*)d_in[3];
  const float* wo = (const float*)d_in[4];
  const float* qw = (const float*)d_in[5];
  const float* kw = (const float*)d_in[6];

  char* ws = (char*)d_ws;
  // workspace layout (bytes), all 256B-aligned offsets
  __bf16* xb    = (__bf16*)(ws + 0);           //  32 MB: x bf16 [M][H]
  __bf16* wqkvT = (__bf16*)(ws + 33554432);    //  48 MB: [6144][4096] bf16
  __bf16* woT   = (__bf16*)(ws + 83886080);    //  32 MB: [4096][4096] bf16
  __bf16* qkv   = (__bf16*)(ws + 117440512);   //  48 MB: [M][6144] bf16
  __bf16* qh    = (__bf16*)(ws + 167772160);   //  32 MB: [b][32][s][128]
  __bf16* kh    = (__bf16*)(ws + 201326592);   //   8 MB: [b][8][s][128]
  __bf16* vT    = (__bf16*)(ws + 209715200);   //   8 MB: [b][8][128][s]
  __bf16* attn  = (__bf16*)(ws + 218103808);   //  32 MB: [M][4096] bf16

  // 1) convert x to bf16
  k_f32_to_bf16<<<(Mrows * (long long)Hc) / 256, 256, 0, stream>>>(
      x, xb, (long long)Mrows * Hc);

  // 2) transpose-convert weights into [N][K] bf16
  k_transpose_bf16<<<((long long)Hc * NQc * Dc) / 256, 256, 0, stream>>>(
      wq, wqkvT, Hc, NQc * Dc);
  k_transpose_bf16<<<((long long)Hc * NKVc * Dc) / 256, 256, 0, stream>>>(
      wk, wqkvT + (long long)NQc * Dc * Hc, Hc, NKVc * Dc);
  k_transpose_bf16<<<((long long)Hc * NKVc * Dc) / 256, 256, 0, stream>>>(
      wv, wqkvT + (long long)(NQc + NKVc) * Dc * Hc, Hc, NKVc * Dc);
  k_transpose_bf16<<<((long long)(NQc * Dc) * Hc) / 256, 256, 0, stream>>>(
      wo, woT, NQc * Dc, Hc);

  // 3) fused QKV projection GEMM: [4096 x 4096] * [4096 x 6144] -> bf16
  dim3 g1(Mrows / 128, QKVN / 128);
  k_gemm<false><<<g1, 256, 0, stream>>>(xb, wqkvT, qkv, Mrows, QKVN, Hc);

  // 4) RMSNorm + RoPE, scatter to head-major q/k and transposed v
  k_normrope<<<Mrows * 48, 128, 0, stream>>>(qkv, qw, kw, qh, kh, vT);

  // 5) causal GQA flash attention
  k_attn<<<Bc * NQc * (Sc / 128), 256, 0, stream>>>(qh, kh, vT, attn);

  // 6) output projection GEMM -> fp32 d_out
  dim3 g2(Mrows / 128, Hc / 128);
  k_gemm<true><<<g2, 256, 0, stream>>>(attn, woT, (float*)d_out, Mrows, Hc,
                                       NQc * Dc);
}
